// Geoattn_GNN_17626545783498
// MI455X (gfx1250) — matrix operand order repaired and checked by reading, not compiled, not run on hardware
//
#include <hip/hip_runtime.h>
#include <math.h>

// ---------------- problem constants ----------------
constexpr int N_NODES = 10000;
constexpr int E_EDGES = 160000;
constexpr int SCA = 256, VEC = 64, HID = 128, EDIM = 64, NET = 4, NG = 60;

// ---------------- WMMA helpers (CDNA5, wave32) ----------------
typedef __attribute__((ext_vector_type(16))) _Float16 v16h;
typedef __attribute__((ext_vector_type(8)))  float    v8f;

// A-matrix (f16 16x32) lane/K mapping per ISA 7.12.2:
// lane<16: K = {0..7, 16..23}; lane>=16: K = {8..15, 24..31}; M = lane&15
__device__ __forceinline__ int akof(int lane, int j) {
  return ((j & 8) << 1) + ((lane & 16) >> 1) + (j & 7);
}

// Load A fragment from LDS f16 tile, row-major [16][kstride]
__device__ __forceinline__ v16h ldsA(const _Float16* A, int kstride, int k0, int lane) {
  v16h a;
  const _Float16* p = A + (lane & 15) * kstride + k0;
#pragma unroll
  for (int j = 0; j < 16; ++j) a[j] = p[akof(lane, j)];
  return a;
}

// Packed-weight B fragment: fragment (nt,ks) = 512 contiguous halves, lane-major,
// so each lane issues one aligned 32B load (lowers to global_load_b128 pair).
__device__ __forceinline__ v16h pkB(const _Float16* __restrict__ P, int ldK, int nt, int ks, int lane) {
  return *(const v16h*)(P + (((size_t)(nt * ldK + ks)) << 9) + ((size_t)lane << 4));
}

__device__ __forceinline__ v8f wmma16(v16h a, v16h b, v8f c) {
  return __builtin_amdgcn_wmma_f32_16x16x32_f16(false, a, false, b, (short)0, c, false, false);
}

// D[16x16] = A(LDS f16 [16][kstride]) x packed-W^T tile nt, summing K
__device__ __forceinline__ v8f gemmKp(const _Float16* Alds, int kstride, int K,
                                      const _Float16* __restrict__ P, int nt, int lane) {
  v8f acc = {};
  int ldK = K >> 5;
  for (int ks = 0; ks < ldK; ++ks)
    acc = wmma16(ldsA(Alds, kstride, ks * 32, lane), pkB(P, ldK, nt, ks, lane), acc);
  return acc;
}
// C/D layout: element r of lane -> M = r + (lane>=16 ? 8 : 0), N = lane&15

// ---------------- weight pre-pack: f32 row-major -> fragment-major f16 ----------------
__global__ void k_pack_w(const float* __restrict__ W, int ldb, int colOff, int Nout, int K,
                         _Float16* __restrict__ P) {
  int ldK = K >> 5;
  size_t total = (size_t)(Nout >> 4) * ldK * 512;
  for (size_t idx = (size_t)blockIdx.x * blockDim.x + threadIdx.x; idx < total;
       idx += (size_t)gridDim.x * blockDim.x) {
    int f = (int)(idx >> 9), r = (int)(idx & 511);
    int lane = r >> 4, h = r & 15;
    int nt = f / ldK, ks = f % ldK;
    int n = nt * 16 + (lane & 15);
    int k = ks * 32 + ((h & 8) << 1) + ((lane & 16) >> 1) + (h & 7);
    P[idx] = (_Float16)W[(size_t)n * ldb + colOff + k];
  }
}

// ---------------- K0: rank-1 edge-vector constants ----------------
__global__ void k_const_vecs(const float* __restrict__ W_ee, const float* __restrict__ em_Wv,
                             const float* __restrict__ em_Wv2, const float* __restrict__ Wev,
                             float* uabs, float* u_ev) {
  __shared__ float ue[VEC], wv[VEC];
  int t = threadIdx.x;
  if (t < VEC) {
    float s = 0.f;
    for (int c = 0; c < EDIM; ++c) s += em_Wv[t * EDIM + c] * W_ee[c];
    ue[t] = s;
    uabs[t] = fabsf(s);
  }
  __syncthreads();
  if (t < VEC) {
    float s = 0.f;
    for (int c = 0; c < VEC; ++c) s += em_Wv2[t * VEC + c] * ue[c];
    wv[t] = s;
  }
  __syncthreads();
  if (t < HID) {
    float s = 0.f;
    for (int c = 0; c < VEC; ++c) s += Wev[t * VEC + c] * wv[c];
    u_ev[t] = s;
  }
}

// ---------------- K1: node mapper + residual connecter (GVLinear x2) ----------------
__global__ __launch_bounds__(128) void k_node_map(
    const float* __restrict__ node_sca, const float* __restrict__ node_vec,
    const _Float16* __restrict__ p_nm_Wv, const _Float16* __restrict__ p_nm_Ws,
    const _Float16* __restrict__ p_nm_Wv2,
    const _Float16* __restrict__ p_rc_Wv, const _Float16* __restrict__ p_rc_Ws,
    const _Float16* __restrict__ p_rc_Wv2,
    float* __restrict__ n_sca_m, float* __restrict__ n_vec_m,
    float* __restrict__ r_sca, float* __restrict__ r_vec) {
  __shared__ _Float16 Avec[3][16][VEC];
  __shared__ _Float16 vi16[3][16][VEC];
  __shared__ _Float16 Asca[16][SCA + VEC];  // [vnorm(64) | sca(256)]
  int tid = threadIdx.x, w = tid >> 5, lane = tid & 31;
  int n0r = blockIdx.x * 16;
  if (n0r >= N_NODES) return;
  for (int idx = tid; idx < 16 * VEC; idx += 128) {
    int m = idx >> 6, c = idx & 63;
    const float* s = node_vec + ((size_t)(n0r + m) * VEC + c) * 3;
    Avec[0][m][c] = (_Float16)s[0];
    Avec[1][m][c] = (_Float16)s[1];
    Avec[2][m][c] = (_Float16)s[2];
  }
  for (int idx = tid; idx < 16 * SCA; idx += 128) {
    int m = idx >> 8, c = idx & 255;
    Asca[m][VEC + c] = (_Float16)node_sca[(size_t)(n0r + m) * SCA + c];
  }
  __syncthreads();
  int mb = (lane & 16) ? 8 : 0, nn = lane & 15;
  for (int pass = 0; pass < 2; ++pass) {
    const _Float16* Wv  = pass ? p_rc_Wv  : p_nm_Wv;
    const _Float16* Ws  = pass ? p_rc_Ws  : p_nm_Ws;
    const _Float16* Wv2 = pass ? p_rc_Wv2 : p_nm_Wv2;
    float* out_s = pass ? r_sca : n_sca_m;
    float* out_v = pass ? r_vec : n_vec_m;
    for (int t = w; t < 12; t += 4) {          // vi = vec @ Wv^T, 3 comps x 4 tiles
      int i = t >> 2, nt = t & 3;
      v8f c = gemmKp(&Avec[i][0][0], VEC, VEC, Wv, nt, lane);
#pragma unroll
      for (int r = 0; r < 8; ++r) vi16[i][mb + r][nt * 16 + nn] = (_Float16)c[r];
    }
    __syncthreads();
    for (int idx = tid; idx < 16 * VEC; idx += 128) {   // vnorm -> Asca[:,0:64]
      int m = idx >> 6, c = idx & 63;
      float x = (float)vi16[0][m][c], y = (float)vi16[1][m][c], z = (float)vi16[2][m][c];
      Asca[m][c] = (_Float16)sqrtf(x * x + y * y + z * z);
    }
    __syncthreads();
    for (int t = w; t < 16; t += 4) {          // out_sca, K=320
      v8f c = gemmKp(&Asca[0][0], SCA + VEC, SCA + VEC, Ws, t, lane);
#pragma unroll
      for (int r = 0; r < 8; ++r)
        out_s[(size_t)(n0r + mb + r) * SCA + t * 16 + nn] = c[r];
    }
    for (int t = w; t < 12; t += 4) {          // out_vec comp-major [N][3][64]
      int i = t >> 2, nt = t & 3;
      v8f c = gemmKp(&vi16[i][0][0], VEC, VEC, Wv2, nt, lane);
#pragma unroll
      for (int r = 0; r < 8; ++r)
        out_v[(size_t)(n0r + mb + r) * (3 * VEC) + i * VEC + nt * 16 + nn] = c[r];
    }
    __syncthreads();
  }
}

// ---------------- K2: node-level precomputes (Ann, Ans, P1, P2, Vnv, Vva) ----------------
__global__ __launch_bounds__(128) void k_node_feats(
    const float* __restrict__ n_sca_m, const float* __restrict__ n_vec_m,
    const _Float16* __restrict__ p_Wnn, const float* __restrict__ bnn,
    const _Float16* __restrict__ p_Wns, const float* __restrict__ bns,
    const _Float16* __restrict__ p_Wsa1, const _Float16* __restrict__ p_Wsa2,
    const float* __restrict__ bsa,
    const _Float16* __restrict__ p_Wnv, const _Float16* __restrict__ p_Wva,
    float* __restrict__ Ann, float* __restrict__ Ans, float* __restrict__ P1,
    float* __restrict__ P2, float* __restrict__ Vnv, float* __restrict__ Vva) {
  __shared__ _Float16 As[16][SCA];
  __shared__ _Float16 Av[3][16][VEC];
  int tid = threadIdx.x, w = tid >> 5, lane = tid & 31;
  int n0r = blockIdx.x * 16;
  if (n0r >= N_NODES) return;
  for (int idx = tid; idx < 16 * SCA; idx += 128) {
    int m = idx >> 8, c = idx & 255;
    As[m][c] = (_Float16)n_sca_m[(size_t)(n0r + m) * SCA + c];
  }
  for (int idx = tid; idx < 16 * VEC; idx += 128) {
    int m = idx >> 6, c = idx & 63;
    const float* s = n_vec_m + (size_t)(n0r + m) * (3 * VEC);
    Av[0][m][c] = (_Float16)s[c];
    Av[1][m][c] = (_Float16)s[VEC + c];
    Av[2][m][c] = (_Float16)s[2 * VEC + c];
  }
  __syncthreads();
  int mb = (lane & 16) ? 8 : 0, nn = lane & 15;
  for (int t = w; t < 32; t += 4) {  // 4 scalar GEMMs x 8 n-tiles (K=256, Nout=128)
    int g = t >> 3, nt = t & 7;
    const _Float16* W = (g == 0) ? p_Wnn : (g == 1) ? p_Wns : (g == 2) ? p_Wsa1 : p_Wsa2;
    const float* bias = (g == 0) ? bnn : (g == 1) ? bns : (g == 2) ? bsa : nullptr;
    float* out = (g == 0) ? Ann : (g == 1) ? Ans : (g == 2) ? P1 : P2;
    v8f c = gemmKp(&As[0][0], SCA, SCA, W, nt, lane);
    float b = bias ? bias[nt * 16 + nn] : 0.f;
#pragma unroll
    for (int r = 0; r < 8; ++r)
      out[(size_t)(n0r + mb + r) * HID + nt * 16 + nn] = c[r] + b;
  }
  for (int t = w; t < 48; t += 4) {  // Vnv/Vva: 2 x 3 comps x 8 tiles (K=64, Nout=128)
    int g = t / 24, rem = t % 24, i = rem >> 3, nt = rem & 7;
    const _Float16* W = g ? p_Wva : p_Wnv;
    float* out = g ? Vva : Vnv;
    v8f c = gemmKp(&Av[i][0][0], VEC, VEC, W, nt, lane);
#pragma unroll
    for (int r = 0; r < 8; ++r)
      out[(size_t)(n0r + mb + r) * (3 * HID) + i * HID + nt * 16 + nn] = c[r];
  }
}

// ---------------- K3: per-edge tile: features -> em_Ws/Wen/Wes + logits + gates ----------------
__global__ __launch_bounds__(128) void k_edge_a(
    const int* __restrict__ eidx, const float* __restrict__ node_pos,
    const float* __restrict__ edge_feature, const _Float16* __restrict__ p_em_Ws,
    const _Float16* __restrict__ p_Wen, const float* __restrict__ ben,
    const _Float16* __restrict__ p_Wes, const float* __restrict__ bes,
    const float* __restrict__ Wsa, const float* __restrict__ P1, const float* __restrict__ P2,
    const float* __restrict__ Vva, const float* __restrict__ uabs,
    float* __restrict__ evn_o, float* __restrict__ En, float* __restrict__ Es,
    float* __restrict__ logits, float* __restrict__ alpha_vec, unsigned int* __restrict__ segmax) {
  __shared__ _Float16 Ae[16][EDIM + VEC];    // [vnorm(64) | gauss(60) | feat(4)]
  __shared__ _Float16 Esca[16][SCA];
  __shared__ float edS[16];
  __shared__ int rS[16], r2S[16], c2S[16];
  __shared__ float dotP[16][8];
  int tid = threadIdx.x, w = tid >> 5, lane = tid & 31;
  int e0 = blockIdx.x * 16;
  if (e0 >= E_EDGES) return;
  if (tid == 0) { __builtin_prefetch(p_em_Ws, 0, 1); __builtin_prefetch(p_Wen, 0, 1); }
  if (tid < 16) {
    int e = e0 + tid;
    int re = eidx[e], ce = eidx[E_EDGES + e];
    rS[tid] = re;
    r2S[tid] = eidx[re];   // n_sca_e[row] pattern -> row[row[e]]
    c2S[tid] = eidx[ce];   // n_sca_e[col] pattern -> row[col[e]]
    float dx = node_pos[re * 3 + 0] - node_pos[ce * 3 + 0];
    float dy = node_pos[re * 3 + 1] - node_pos[ce * 3 + 1];
    float dz = node_pos[re * 3 + 2] - node_pos[ce * 3 + 2];
    float d = sqrtf(dx * dx + dy * dy + dz * dz);
    edS[tid] = d;
    float rinv = 1.f / (d + 1e-7f);
    evn_o[e * 3 + 0] = dx * rinv;
    evn_o[e * 3 + 1] = dy * rinv;
    evn_o[e * 3 + 2] = dz * rinv;
  }
  __syncthreads();
  const float gstep = 10.0f / (NG - 1);
  const float gcoef = -0.5f / (gstep * gstep);
  for (int idx = tid; idx < 16 * (EDIM + VEC); idx += 128) {
    int m = idx >> 7, c = idx & 127;
    float d = edS[m], v;
    if (c < VEC) {
      v = uabs[c] * (d / (d + 1e-7f));       // rank-1 vnorm of mapped edge vec
    } else if (c < VEC + NG) {
      float x = d - gstep * (c - VEC);
      v = __expf(gcoef * x * x);
    } else {
      v = edge_feature[(size_t)(e0 + m) * NET + (c - VEC - NG)];
    }
    Ae[m][c] = (_Float16)v;
  }
  __syncthreads();
  int mb = (lane & 16) ? 8 : 0, nn = lane & 15;
  for (int t = w; t < 16; t += 4) {          // e_sca_m: K=128, Nout=256 (kept in LDS f16)
    v8f c = gemmKp(&Ae[0][0], EDIM + VEC, EDIM + VEC, p_em_Ws, t, lane);
#pragma unroll
    for (int r = 0; r < 8; ++r) Esca[mb + r][t * 16 + nn] = (_Float16)c[r];
  }
  __syncthreads();
  for (int t = w; t < 16; t += 4) {          // En / Es: K=256, Nout=128
    int g = t >> 3, nt = t & 7;
    const _Float16* W = g ? p_Wes : p_Wen;
    const float* bias = g ? bes : ben;
    float* out = g ? Es : En;
    v8f c = gemmKp(&Esca[0][0], SCA, SCA, W, nt, lane);
    float b = bias[nt * 16 + nn];
#pragma unroll
    for (int r = 0; r < 8; ++r)
      out[(size_t)(e0 + mb + r) * HID + nt * 16 + nn] = c[r] + b;
  }
  // attention logits (P1 already carries bsa) + ordered-uint atomic segment max
  for (int idx = tid; idx < 16 * HID; idx += 128) {
    int m = idx >> 7, h = idx & 127;
    float lg = P1[(size_t)r2S[m] * HID + h] + P2[(size_t)c2S[m] * HID + h]
             + edS[m] * Wsa[(size_t)h * (2 * SCA + 1) + 2 * SCA];
    logits[(size_t)(e0 + m) * HID + h] = lg;
    unsigned int bits = __float_as_uint(lg);
    unsigned int key = (bits & 0x80000000u) ? ~bits : (bits | 0x80000000u);
    atomicMax(&segmax[(size_t)rS[m] * HID + h], key);
  }
  // vector gate: sigmoid( sum Vva[r2] . Vva[c2] )
  {
    int m = tid & 15, part = tid >> 4;
    const float* a = Vva + (size_t)r2S[m] * (3 * HID);
    const float* b = Vva + (size_t)c2S[m] * (3 * HID);
    float s = 0.f;
    for (int c = part * 48; c < part * 48 + 48; ++c) s += a[c] * b[c];
    dotP[m][part] = s;
  }
  __syncthreads();
  if (tid < 16) {
    float s = 0.f;
#pragma unroll
    for (int p = 0; p < 8; ++p) s += dotP[tid][p];
    alpha_vec[e0 + tid] = 1.f / (1.f + __expf(-s));
  }
}

// ---------------- K4: softmax exp + segment sum ----------------
__global__ void k_softmax_norm(const int* __restrict__ eidx, const unsigned int* __restrict__ segmax,
                               float* __restrict__ logits, float* __restrict__ segsum) {
  size_t i = (size_t)blockIdx.x * blockDim.x + threadIdx.x;
  if (i >= (size_t)E_EDGES * HID) return;
  int e = (int)(i >> 7), h = (int)(i & 127);
  int re = eidx[e];
  unsigned int key = segmax[(size_t)re * HID + h];
  unsigned int bits = (key & 0x80000000u) ? (key & 0x7fffffffu) : ~key;
  float ex = __expf(logits[i] - __uint_as_float(bits));
  logits[i] = ex;
  atomicAdd(&segsum[(size_t)re * HID + h], ex);
}

// ---------------- K5: per-edge messages + msg_out GVLinear + aggregation ----------------
__global__ __launch_bounds__(128) void k_edge_msg(
    const int* __restrict__ eidx, const float* __restrict__ logits, const float* __restrict__ segsum,
    const float* __restrict__ En, const float* __restrict__ Es,
    const float* __restrict__ Ann, const float* __restrict__ Ans, const float* __restrict__ Vnv,
    const float* __restrict__ evn, const float* __restrict__ alpha_vec, const float* __restrict__ u_ev,
    const _Float16* __restrict__ p_mo_Wv, const _Float16* __restrict__ p_mo_Ws,
    const _Float16* __restrict__ p_mo_Wv2,
    float* __restrict__ aggr_sca, float* __restrict__ aggr_vec) {
  __shared__ _Float16 Mvec[3][16][HID];
  __shared__ _Float16 Vi[3][16][HID];
  __shared__ _Float16 A2[16][2 * HID];   // [vnorm(128) | msg_sca(128)]
  __shared__ int rS[16], r2S[16];
  __shared__ float avS[16], evS[16][3];
  int tid = threadIdx.x, w = tid >> 5, lane = tid & 31;
  int e0 = blockIdx.x * 16;
  if (e0 >= E_EDGES) return;
  if (tid == 0) { __builtin_prefetch(p_mo_Wv, 0, 1); __builtin_prefetch(p_mo_Ws, 0, 1); }
  if (tid < 16) {
    int e = e0 + tid;
    int re = eidx[e];
    rS[tid] = re;
    r2S[tid] = eidx[re];
    avS[tid] = alpha_vec[e];
    evS[tid][0] = evn[e * 3 + 0];
    evS[tid][1] = evn[e * 3 + 1];
    evS[tid][2] = evn[e * 3 + 2];
  }
  __syncthreads();
  for (int idx = tid; idx < 16 * HID; idx += 128) {
    int m = idx >> 7, h = idx & 127;
    size_t eoff = (size_t)(e0 + m) * HID + h;
    int r2 = r2S[m];
    float alpha = logits[eoff] / segsum[(size_t)rS[m] * HID + h];
    float msca = Ann[(size_t)r2 * HID + h] * En[eoff] * alpha;
    A2[m][HID + h] = (_Float16)msca;
    float av = avS[m];
    float ansu = Ans[(size_t)r2 * HID + h] * u_ev[h];   // edge_vec_hid is rank-1
    float es = Es[eoff];
    const float* vnv = Vnv + (size_t)r2 * (3 * HID) + h;
#pragma unroll
    for (int i = 0; i < 3; ++i)
      Mvec[i][m][h] = (_Float16)(av * (ansu * evS[m][i] + vnv[i * HID] * es));
  }
  __syncthreads();
  int mb = (lane & 16) ? 8 : 0, nn = lane & 15;
  for (int t = w; t < 24; t += 4) {      // vi = msg_vec @ mo_Wv^T (K=128, Nout=128) x3
    int i = t >> 3, nt = t & 7;
    v8f c = gemmKp(&Mvec[i][0][0], HID, HID, p_mo_Wv, nt, lane);
#pragma unroll
    for (int r = 0; r < 8; ++r) Vi[i][mb + r][nt * 16 + nn] = (_Float16)c[r];
  }
  __syncthreads();
  for (int idx = tid; idx < 16 * HID; idx += 128) {
    int m = idx >> 7, h = idx & 127;
    float x = (float)Vi[0][m][h], y = (float)Vi[1][m][h], z = (float)Vi[2][m][h];
    A2[m][h] = (_Float16)sqrtf(x * x + y * y + z * z);
  }
  __syncthreads();
  for (int t = w; t < 16; t += 4) {      // msg_sca out: K=256, Nout=256 -> atomic segment sum
    v8f c = gemmKp(&A2[0][0], 2 * HID, 2 * HID, p_mo_Ws, t, lane);
#pragma unroll
    for (int r = 0; r < 8; ++r)
      atomicAdd(&aggr_sca[(size_t)rS[mb + r] * SCA + t * 16 + nn], c[r]);
  }
  for (int t = w; t < 12; t += 4) {      // msg_vec out: K=128, Nout=64 x3 -> atomic segment sum
    int i = t >> 2, nt = t & 3;
    v8f c = gemmKp(&Vi[i][0][0], HID, HID, p_mo_Wv2, nt, lane);
#pragma unroll
    for (int r = 0; r < 8; ++r)
      atomicAdd(&aggr_vec[(size_t)rS[mb + r] * (3 * VEC) + i * VEC + nt * 16 + nn], c[r]);
  }
}

// ---------------- K6: aggr_out GVPerceptron + VN-LeakyReLU + LeakyReLU ----------------
__global__ __launch_bounds__(128) void k_node_out(
    const float* __restrict__ r_sca, const float* __restrict__ r_vec,
    const float* __restrict__ aggr_sca, const float* __restrict__ aggr_vec,
    const _Float16* __restrict__ p_ao_Wv, const _Float16* __restrict__ p_ao_Ws,
    const _Float16* __restrict__ p_ao_Wv2, const _Float16* __restrict__ p_ao_Wd,
    float* __restrict__ out) {
  __shared__ _Float16 Av[3][16][VEC];
  __shared__ _Float16 A[16][SCA + VEC];
  __shared__ _Float16 vi16[3][16][VEC];
  __shared__ float ov[3][16][VEC];
  __shared__ _Float16 ov16[3][16][VEC];
  __shared__ float dd[3][16][VEC];
  int tid = threadIdx.x, w = tid >> 5, lane = tid & 31;
  int n0r = blockIdx.x * 16;
  if (n0r >= N_NODES) return;
  for (int idx = tid; idx < 16 * SCA; idx += 128) {
    int m = idx >> 8, c = idx & 255;
    size_t o = (size_t)(n0r + m) * SCA + c;
    A[m][VEC + c] = (_Float16)(r_sca[o] + aggr_sca[o]);
  }
  for (int idx = tid; idx < 16 * VEC; idx += 128) {
    int m = idx >> 6, c = idx & 63;
    size_t o = (size_t)(n0r + m) * (3 * VEC) + c;
#pragma unroll
    for (int i = 0; i < 3; ++i)
      Av[i][m][c] = (_Float16)(r_vec[o + i * VEC] + aggr_vec[o + i * VEC]);
  }
  __syncthreads();
  int mb = (lane & 16) ? 8 : 0, nn = lane & 15;
  for (int t = w; t < 12; t += 4) {
    int i = t >> 2, nt = t & 3;
    v8f c = gemmKp(&Av[i][0][0], VEC, VEC, p_ao_Wv, nt, lane);
#pragma unroll
    for (int r = 0; r < 8; ++r) vi16[i][mb + r][nt * 16 + nn] = (_Float16)c[r];
  }
  __syncthreads();
  for (int idx = tid; idx < 16 * VEC; idx += 128) {
    int m = idx >> 6, c = idx & 63;
    float x = (float)vi16[0][m][c], y = (float)vi16[1][m][c], z = (float)vi16[2][m][c];
    A[m][c] = (_Float16)sqrtf(x * x + y * y + z * z);
  }
  __syncthreads();
  for (int t = w; t < 16; t += 4) {      // out_sca (K=320) + LeakyReLU(0.01)
    v8f c = gemmKp(&A[0][0], SCA + VEC, SCA + VEC, p_ao_Ws, t, lane);
#pragma unroll
    for (int r = 0; r < 8; ++r) {
      float v = c[r];
      out[(size_t)(n0r + mb + r) * SCA + t * 16 + nn] = (v >= 0.f) ? v : 0.01f * v;
    }
  }
  for (int t = w; t < 12; t += 4) {      // out_vec pre-VN
    int i = t >> 2, nt = t & 3;
    v8f c = gemmKp(&vi16[i][0][0], VEC, VEC, p_ao_Wv2, nt, lane);
#pragma unroll
    for (int r = 0; r < 8; ++r) {
      ov[i][mb + r][nt * 16 + nn] = c[r];
      ov16[i][mb + r][nt * 16 + nn] = (_Float16)c[r];
    }
  }
  __syncthreads();
  for (int t = w; t < 12; t += 4) {      // d = ov @ ao_Wd^T
    int i = t >> 2, nt = t & 3;
    v8f c = gemmKp(&ov16[i][0][0], VEC, VEC, p_ao_Wd, nt, lane);
#pragma unroll
    for (int r = 0; r < 8; ++r) dd[i][mb + r][nt * 16 + nn] = c[r];
  }
  __syncthreads();
  float* out_vec = out + (size_t)N_NODES * SCA;
  for (int idx = tid; idx < 16 * VEC; idx += 128) {
    int m = idx >> 6, c = idx & 63;
    float vx = ov[0][m][c], vy = ov[1][m][c], vz = ov[2][m][c];
    float dx = dd[0][m][c], dy = dd[1][m][c], dz = dd[2][m][c];
    float dot = vx * dx + vy * dy + vz * dz;
    float dns = dx * dx + dy * dy + dz * dz;
    float s = (dot >= 0.f) ? 0.f : dot / (dns + 1e-6f);
    float wx = vx - s * dx, wy = vy - s * dy, wz = vz - s * dz;
    size_t o = ((size_t)(n0r + m) * VEC + c) * 3;
    out_vec[o + 0] = 0.2f * vx + 0.8f * wx;
    out_vec[o + 1] = 0.2f * vy + 0.8f * wy;
    out_vec[o + 2] = 0.2f * vz + 0.8f * wz;
  }
}

// ---------------- host launcher ----------------
extern "C" void kernel_launch(void* const* d_in, const int* in_sizes, int n_in,
                              void* d_out, int out_size, void* d_ws, size_t ws_size,
                              hipStream_t stream) {
  (void)in_sizes; (void)n_in; (void)out_size; (void)ws_size;
  const float* node_sca = (const float*)d_in[0];
  const float* node_vec = (const float*)d_in[1];
  const float* node_pos = (const float*)d_in[2];
  const float* edge_feature = (const float*)d_in[3];
  const float* W_ee = (const float*)d_in[4];
  const float* nm_Wv = (const float*)d_in[5];
  const float* nm_Ws = (const float*)d_in[6];
  const float* nm_Wv2 = (const float*)d_in[7];
  const float* em_Wv = (const float*)d_in[8];
  const float* em_Ws = (const float*)d_in[9];
  const float* em_Wv2 = (const float*)d_in[10];
  const float* Wen = (const float*)d_in[11];
  const float* ben = (const float*)d_in[12];
  const float* Wnn = (const float*)d_in[13];
  const float* bnn = (const float*)d_in[14];
  const float* Wes = (const float*)d_in[15];
  const float* bes = (const float*)d_in[16];
  const float* Wns = (const float*)d_in[17];
  const float* bns = (const float*)d_in[18];
  const float* Wev = (const float*)d_in[19];
  const float* Wnv = (const float*)d_in[20];
  const float* Wsa = (const float*)d_in[21];
  const float* bsa = (const float*)d_in[22];
  const float* Wva = (const float*)d_in[23];
  const float* mo_Wv = (const float*)d_in[24];
  const float* mo_Ws = (const float*)d_in[25];
  const float* mo_Wv2 = (const float*)d_in[26];
  const float* rc_Wv = (const float*)d_in[27];
  const float* rc_Ws = (const float*)d_in[28];
  const float* rc_Wv2 = (const float*)d_in[29];
  const float* ao_Wv = (const float*)d_in[30];
  const float* ao_Ws = (const float*)d_in[31];
  const float* ao_Wv2 = (const float*)d_in[32];
  const float* ao_Wd = (const float*)d_in[33];
  const int* eidx = (const int*)d_in[34];

  float* W = (float*)d_ws;
  size_t off = 0;
  auto alloc = [&](size_t n) { float* p = W + off; off += n; return p; };
  auto allocH = [&](size_t halves) { return (_Float16*)alloc(halves / 2); };

  // packed f16 weights (fragment-major); each size = Nout*K halves
  _Float16* p_nm_Wv  = allocH(64 * 64);
  _Float16* p_nm_Ws  = allocH(256 * 320);
  _Float16* p_nm_Wv2 = allocH(64 * 64);
  _Float16* p_rc_Wv  = allocH(64 * 64);
  _Float16* p_rc_Ws  = allocH(256 * 320);
  _Float16* p_rc_Wv2 = allocH(64 * 64);
  _Float16* p_em_Ws  = allocH(256 * 128);
  _Float16* p_Wen    = allocH(128 * 256);
  _Float16* p_Wes    = allocH(128 * 256);
  _Float16* p_Wnn    = allocH(128 * 256);
  _Float16* p_Wns    = allocH(128 * 256);
  _Float16* p_Wsa1   = allocH(128 * 256);
  _Float16* p_Wsa2   = allocH(128 * 256);
  _Float16* p_Wnv    = allocH(128 * 64);
  _Float16* p_Wva    = allocH(128 * 64);
  _Float16* p_mo_Wv  = allocH(128 * 128);
  _Float16* p_mo_Ws  = allocH(256 * 256);
  _Float16* p_mo_Wv2 = allocH(64 * 128);
  _Float16* p_ao_Wv  = allocH(64 * 64);
  _Float16* p_ao_Ws  = allocH(256 * 320);
  _Float16* p_ao_Wv2 = allocH(64 * 64);
  _Float16* p_ao_Wd  = allocH(64 * 64);

  float* n_sca_m = alloc((size_t)N_NODES * SCA);
  float* n_vec_m = alloc((size_t)N_NODES * 3 * VEC);
  float* r_sca   = alloc((size_t)N_NODES * SCA);
  float* r_vec   = alloc((size_t)N_NODES * 3 * VEC);
  float* Ann     = alloc((size_t)N_NODES * HID);
  float* Ans     = alloc((size_t)N_NODES * HID);
  float* P1      = alloc((size_t)N_NODES * HID);
  float* P2      = alloc((size_t)N_NODES * HID);
  float* Vnv     = alloc((size_t)N_NODES * 3 * HID);
  float* Vva     = alloc((size_t)N_NODES * 3 * HID);
  float* uabs    = alloc(VEC);
  float* u_ev    = alloc(HID);
  float* evn     = alloc((size_t)E_EDGES * 3);
  float* En      = alloc((size_t)E_EDGES * HID);
  float* Es      = alloc((size_t)E_EDGES * HID);
  float* logits  = alloc((size_t)E_EDGES * HID);
  float* alpha_v = alloc((size_t)E_EDGES);
  float* segsum  = alloc((size_t)N_NODES * HID);
  float* aggr_s  = alloc((size_t)N_NODES * SCA);
  float* aggr_v  = alloc((size_t)N_NODES * 3 * VEC);
  unsigned int* segmax = (unsigned int*)alloc((size_t)N_NODES * HID);

  hipMemsetAsync(segmax, 0, (size_t)N_NODES * HID * sizeof(unsigned int), stream);
  hipMemsetAsync(segsum, 0, (size_t)N_NODES * HID * sizeof(float), stream);
  hipMemsetAsync(aggr_s, 0, (size_t)N_NODES * SCA * sizeof(float), stream);
  hipMemsetAsync(aggr_v, 0, (size_t)N_NODES * 3 * VEC * sizeof(float), stream);

  // pack all weights to fragment-major f16
  auto pack = [&](const float* src, int ldb, int colOff, int Nout, int K, _Float16* dst) {
    int total = Nout * K;
    int grid = (total + 255) / 256;
    k_pack_w<<<grid, 256, 0, stream>>>(src, ldb, colOff, Nout, K, dst);
  };
  pack(nm_Wv, 64, 0, 64, 64, p_nm_Wv);
  pack(nm_Ws, 320, 0, 256, 320, p_nm_Ws);
  pack(nm_Wv2, 64, 0, 64, 64, p_nm_Wv2);
  pack(rc_Wv, 64, 0, 64, 64, p_rc_Wv);
  pack(rc_Ws, 320, 0, 256, 320, p_rc_Ws);
  pack(rc_Wv2, 64, 0, 64, 64, p_rc_Wv2);
  pack(em_Ws, 128, 0, 256, 128, p_em_Ws);
  pack(Wen, 256, 0, 128, 256, p_Wen);
  pack(Wes, 256, 0, 128, 256, p_Wes);
  pack(Wnn, 256, 0, 128, 256, p_Wnn);
  pack(Wns, 256, 0, 128, 256, p_Wns);
  pack(Wsa, 513, 0, 128, 256, p_Wsa1);
  pack(Wsa, 513, 256, 128, 256, p_Wsa2);
  pack(Wnv, 64, 0, 128, 64, p_Wnv);
  pack(Wva, 64, 0, 128, 64, p_Wva);
  pack(mo_Wv, 128, 0, 128, 128, p_mo_Wv);
  pack(mo_Ws, 256, 0, 256, 256, p_mo_Ws);
  pack(mo_Wv2, 128, 0, 64, 128, p_mo_Wv2);
  pack(ao_Wv, 64, 0, 64, 64, p_ao_Wv);
  pack(ao_Ws, 320, 0, 256, 320, p_ao_Ws);
  pack(ao_Wv2, 64, 0, 64, 64, p_ao_Wv2);
  pack(ao_Wd, 64, 0, 64, 64, p_ao_Wd);

  k_const_vecs<<<1, 128, 0, stream>>>(W_ee, em_Wv, em_Wv2, Wev, uabs, u_ev);
  k_node_map<<<N_NODES / 16, 128, 0, stream>>>(node_sca, node_vec, p_nm_Wv, p_nm_Ws, p_nm_Wv2,
                                               p_rc_Wv, p_rc_Ws, p_rc_Wv2, n_sca_m, n_vec_m, r_sca, r_vec);
  k_node_feats<<<N_NODES / 16, 128, 0, stream>>>(n_sca_m, n_vec_m, p_Wnn, bnn, p_Wns, bns,
                                                 p_Wsa1, p_Wsa2, bsa, p_Wnv, p_Wva,
                                                 Ann, Ans, P1, P2, Vnv, Vva);
  k_edge_a<<<E_EDGES / 16, 128, 0, stream>>>(eidx, node_pos, edge_feature, p_em_Ws, p_Wen, ben,
                                             p_Wes, bes, Wsa, P1, P2, Vva, uabs,
                                             evn, En, Es, logits, alpha_v, segmax);
  k_softmax_norm<<<(int)(((size_t)E_EDGES * HID + 255) / 256), 256, 0, stream>>>(eidx, segmax, logits, segsum);
  k_edge_msg<<<E_EDGES / 16, 128, 0, stream>>>(eidx, logits, segsum, En, Es, Ann, Ans, Vnv,
                                               evn, alpha_v, u_ev, p_mo_Wv, p_mo_Ws, p_mo_Wv2,
                                               aggr_s, aggr_v);
  k_node_out<<<N_NODES / 16, 128, 0, stream>>>(r_sca, r_vec, aggr_s, aggr_v,
                                               p_ao_Wv, p_ao_Ws, p_ao_Wv2, p_ao_Wd, (float*)d_out);
}
